// GroupedProjectedHead_71339406786678
// MI455X (gfx1250) — matrix operand order repaired and checked by reading, not compiled, not run on hardware
//
#include <hip/hip_runtime.h>
#include <hip/hip_bf16.h>

// GroupedProjectedHead for MI455X (gfx1250, wave32, WMMA + TDM).
// Pre-pass: z->bf16, W1->bf16^T [N,K], W2->bf16^T [N,K], Wv->bf16 (already [N,K]).
// Pipeline: GEMM1(zb@W1t+b1 -> bf16) -> LN+GELU(inplace) -> GEMM2(hb@W2t+b2 -> f32)
//           -> q row L2-normalize (bf16) -> Wv row-factor -> GEMM3(qb@Wvb*factor+bv -> out)
// GEMM tiles are staged into LDS by the Tensor Data Mover (tensor_load_to_lds,
// TENSORcnt, double-buffered), consumed by v_wmma_f32_16x16x32_bf16.

#define B_DIM   4096
#define G_DIM   16
#define IN_D    1024
#define HID_D   2048
#define PROJ_D  256
#define CHUNK_D 4096

typedef __attribute__((ext_vector_type(16))) __bf16        v16bf;
typedef __attribute__((ext_vector_type(8)))  float         v8f;
typedef __attribute__((ext_vector_type(4)))  unsigned int  v4u;
typedef __attribute__((ext_vector_type(8)))  int           v8i;
typedef __attribute__((ext_vector_type(4)))  int           v4i;

union FragBF { v16bf v; unsigned int u[8]; };

__device__ __forceinline__ unsigned short f2bf(float f) {
    unsigned int u = __float_as_uint(f);
    return (unsigned short)((u + 0x7FFFu + ((u >> 16) & 1u)) >> 16);
}
__device__ __forceinline__ float bf2f(unsigned short h) {
    return __uint_as_float(((unsigned int)h) << 16);
}

#define KSTEP  32
#define APITCH 34   // 32 bf16 (64B) + 1 DWORD pad per row => 68B pitch, u32-aligned

#if defined(__has_builtin)
#  if __has_builtin(__builtin_amdgcn_tensor_load_to_lds)
#    define USE_TDM 1
#  endif
#endif
#ifndef USE_TDM
#  define USE_TDM 0
#endif

#if USE_TDM
// Issue one TDM 2D tile load: 128 rows x 32 bf16, row pitch = rowStride elements,
// LDS gets 1 DWORD pad per 16 DWORDs (64B row) -> APITCH layout.
// amdgpu-toolchain builtin arity: (v4u, v8i, v4i, v4i, v8i, i32 cpol).
__device__ __forceinline__ void tdm_load_tile(unsigned lds_off,
                                              const unsigned short* gptr,
                                              int dim0, int rowStride)
{
    const unsigned long long ga = (unsigned long long)(const void*)gptr;
    v4u g0; v8i g1;
    v4i g2 = {0, 0, 0, 0};
    v4i g3 = {0, 0, 0, 0};
    v8i g4 = {0, 0, 0, 0, 0, 0, 0, 0};
    g0[0] = 1u;                                        // count=1, user mode
    g0[1] = lds_off;                                   // lds_addr [63:32]
    g0[2] = (unsigned)ga;                              // global_addr lo
    g0[3] = (unsigned)((ga >> 32) & 0x01FFFFFFu)       // global_addr hi (57-bit)
          | (2u << 30);                                // type=2 (image)
    g1[0] = (1 << 16)                                  // data_size = 2 bytes
          | (1 << 20)                                  // pad_enable
          | (3 << 22);                                 // pad_interval: 16 DWORDs
                                                       // pad_amount=0 -> 1 DWORD
    g1[1] = (int)(((unsigned)dim0 & 0xFFFFu) << 16);   // tensor_dim0 lo16
    g1[2] = (int)((((unsigned)dim0 >> 16) & 0xFFFFu)   // tensor_dim0 hi16
          | (128u << 16));                             // tensor_dim1 lo16 = 128
    g1[3] = (int)(32u << 16);                          // tensor_dim1 hi=0, tile_dim0=32
    g1[4] = 128;                                       // tile_dim1=128, tile_dim2=0
    g1[5] = rowStride;                                 // tensor_dim0_stride lo32
    g1[6] = 0;                                         // stride hi, dim1_stride lo
    g1[7] = 0;
    __builtin_amdgcn_tensor_load_to_lds(g0, g1, g2, g3, g4, 0);
}
#endif

// ---------------------------------------------------------------------------
// Tiled WMMA GEMM: C[g] (MxN) = A[g] (M x K bf16, K contiguous) * B[g]^T
// where B[g] is stored N x K bf16 (K contiguous). Block tile 128x128,
// 8 waves (2M x 4N), each wave 4x2 WMMA 16x16 tiles, K-step 32.
// EPI: 0 = +E0[n], store bf16 ; 1 = +E0[n], store f32 ; 2 = *E0[n]+E1[n], f32
// ---------------------------------------------------------------------------
template <int EPI>
__global__ __launch_bounds__(256) void wmma_gemm_kernel(
    const unsigned short* __restrict__ A, const unsigned short* __restrict__ Bw,
    void* __restrict__ C,
    const float* __restrict__ E0, const float* __restrict__ E1,
    int K, int ldc, int N, long sAg, long sBg, long sCg)
{
    __shared__ unsigned short As[2][128 * APITCH];
    __shared__ unsigned short Bs[2][128 * APITCH];

    const int g    = blockIdx.z;
    const int m0   = blockIdx.y * 128;
    const int n0   = blockIdx.x * 128;
    const int tid  = threadIdx.x;
    const int lane = tid & 31;
    const int waveM = (tid >> 5) >> 2;   // 0..1
    const int waveN = (tid >> 5) & 3;    // 0..3
    const int mlane = lane & 15;

    const unsigned short* Ag = A  + (size_t)g * sAg + (size_t)m0 * K;
    const unsigned short* Bg = Bw + (size_t)g * sBg + (size_t)n0 * K;

    v8f acc[4][2];
    for (int i = 0; i < 4; i++)
        for (int j = 0; j < 2; j++)
            for (int e = 0; e < 8; e++) acc[i][j][e] = 0.0f;

    // Per-lane K offsets for fragments (ISA 7.12.2, 16-bit matrices).
    int kkA[8], kkB[8];
    {
        const int abase = (lane >> 4) * 8;    // A: halves interleave K 0-7 / 16-23
        const int bbase = (lane >> 4) * 8;    // B: halves interleave K 0-7 / 16-23
        for (int v = 0; v < 8; v++) {
            kkA[v] = abase + ((v < 4) ? (2 * v) : (16 + 2 * (v - 4)));
            kkB[v] = bbase + ((v < 4) ? (2 * v) : (16 + 2 * (v - 4)));
        }
    }

    const int nk = K / KSTEP;

#if USE_TDM
    const int widu = __builtin_amdgcn_readfirstlane(tid >> 5);  // scalar wave id
    const unsigned offA0 = (unsigned)(unsigned long long)(void*)&As[0][0];
    const unsigned offA1 = (unsigned)(unsigned long long)(void*)&As[1][0];
    const unsigned offB0 = (unsigned)(unsigned long long)(void*)&Bs[0][0];
    const unsigned offB1 = (unsigned)(unsigned long long)(void*)&Bs[1][0];

    if (widu == 0) {
        tdm_load_tile(offA0, Ag, K, K);
        tdm_load_tile(offB0, Bg, K, K);
    }
    for (int kt = 0; kt < nk; kt++) {
        const int cur = kt & 1;
        if (widu == 0) {
            if (kt + 1 < nk) {
                const int kn = (kt + 1) * KSTEP;
                tdm_load_tile(cur ? offA0 : offA1, Ag + kn, K, K);
                tdm_load_tile(cur ? offB0 : offB1, Bg + kn, K, K);
                __builtin_amdgcn_s_wait_tensorcnt(2);  // current pair done
            } else {
                __builtin_amdgcn_s_wait_tensorcnt(0);
            }
        }
        __syncthreads();   // LDS buffer `cur` is ready for everyone

        FragBF afr[4], bfr[2];
        for (int i = 0; i < 4; i++) {
            const unsigned short* rp = &As[cur][(waveM * 64 + i * 16 + mlane) * APITCH];
            for (int v = 0; v < 8; v++) afr[i].u[v] = *(const unsigned int*)(rp + kkA[v]);
        }
        for (int j = 0; j < 2; j++) {
            const unsigned short* rp = &Bs[cur][(waveN * 32 + j * 16 + mlane) * APITCH];
            for (int v = 0; v < 8; v++) bfr[j].u[v] = *(const unsigned int*)(rp + kkB[v]);
        }
        for (int i = 0; i < 4; i++)
            for (int j = 0; j < 2; j++)
                acc[i][j] = __builtin_amdgcn_wmma_f32_16x16x32_bf16(
                    false, afr[i].v, false, bfr[j].v, (short)0, acc[i][j], false, false);
        __syncthreads();   // all reads of `cur` done before TDM overwrites it
    }
#else
    // Fallback: manual global->LDS staging (both operands bf16, K contiguous).
    for (int kt = 0; kt < nk; kt++) {
        const int k0 = kt * KSTEP;
        for (int it = 0; it < 2; it++) {
            const int idx = tid + 256 * it;       // 512 x uint4 (8 bf16)
            const int r = idx >> 2;
            const int c = (idx & 3) * 8;
            const uint4 va = *(const uint4*)(Ag + (size_t)r * K + k0 + c);
            unsigned int* da = (unsigned int*)&As[0][r * APITCH + c];
            da[0] = va.x; da[1] = va.y; da[2] = va.z; da[3] = va.w;
            const uint4 vb = *(const uint4*)(Bg + (size_t)r * K + k0 + c);
            unsigned int* db = (unsigned int*)&Bs[0][r * APITCH + c];
            db[0] = vb.x; db[1] = vb.y; db[2] = vb.z; db[3] = vb.w;
        }
        __syncthreads();
        FragBF afr[4], bfr[2];
        for (int i = 0; i < 4; i++) {
            const unsigned short* rp = &As[0][(waveM * 64 + i * 16 + mlane) * APITCH];
            for (int v = 0; v < 8; v++) afr[i].u[v] = *(const unsigned int*)(rp + kkA[v]);
        }
        for (int j = 0; j < 2; j++) {
            const unsigned short* rp = &Bs[0][(waveN * 32 + j * 16 + mlane) * APITCH];
            for (int v = 0; v < 8; v++) bfr[j].u[v] = *(const unsigned int*)(rp + kkB[v]);
        }
        for (int i = 0; i < 4; i++)
            for (int j = 0; j < 2; j++)
                acc[i][j] = __builtin_amdgcn_wmma_f32_16x16x32_bf16(
                    false, afr[i].v, false, bfr[j].v, (short)0, acc[i][j], false, false);
        __syncthreads();
    }
#endif

    // ---- epilogue ----
    const int mBase = m0 + waveM * 64 + ((lane >> 4) * 8);
    const int nBase = n0 + waveN * 32 + mlane;
    if (EPI == 0) {
        unsigned short* Cg = (unsigned short*)C + (size_t)g * sCg;
        const float* bias = E0 + (size_t)g * N;
        for (int i = 0; i < 4; i++)
            for (int j = 0; j < 2; j++) {
                const int n = nBase + j * 16;
                const float bn = bias[n];
                for (int r = 0; r < 8; r++)
                    Cg[(size_t)(mBase + i * 16 + r) * ldc + n] = f2bf(acc[i][j][r] + bn);
            }
    } else if (EPI == 1) {
        float* Cg = (float*)C + (size_t)g * sCg;
        const float* bias = E0 + (size_t)g * N;
        for (int i = 0; i < 4; i++)
            for (int j = 0; j < 2; j++) {
                const int n = nBase + j * 16;
                const float bn = bias[n];
                for (int r = 0; r < 8; r++)
                    Cg[(size_t)(mBase + i * 16 + r) * ldc + n] = acc[i][j][r] + bn;
            }
    } else {
        float* Cg = (float*)C + (size_t)g * sCg;     // sCg = column offset (CHUNK)
        const float* fac = E0 + (size_t)g * N;
        const float* bvv = E1 + (size_t)g * N;
        for (int i = 0; i < 4; i++)
            for (int j = 0; j < 2; j++) {
                const int n = nBase + j * 16;
                const float f = fac[n], bb = bvv[n];
                for (int r = 0; r < 8; r++)
                    Cg[(size_t)(mBase + i * 16 + r) * ldc + n] = acc[i][j][r] * f + bb;
            }
    }
}

// ---------------------------------------------------------------------------
// Elementwise f32 -> bf16 (4 per thread).
// ---------------------------------------------------------------------------
__global__ __launch_bounds__(256) void cvt_bf16_kernel(
    const float* __restrict__ src, unsigned short* __restrict__ dst)
{
    const long i = (long)blockIdx.x * 256 + threadIdx.x;
    const float4 v = ((const float4*)src)[i];
    const unsigned o0 = (unsigned)f2bf(v.x) | ((unsigned)f2bf(v.y) << 16);
    const unsigned o1 = (unsigned)f2bf(v.z) | ((unsigned)f2bf(v.w) << 16);
    ((uint2*)dst)[i] = make_uint2(o0, o1);
}

// ---------------------------------------------------------------------------
// Per-group 32x32 tiled transpose+convert: src f32 [R,Cn] -> dst bf16 [Cn,R].
// ---------------------------------------------------------------------------
__global__ __launch_bounds__(256) void transpose_bf16_kernel(
    const float* __restrict__ src, unsigned short* __restrict__ dst, int R, int Cn)
{
    __shared__ unsigned short t[32][33];
    const int g  = blockIdx.z;
    const int r0 = blockIdx.x * 32;
    const int c0 = blockIdx.y * 32;
    const float*    S = src + (size_t)g * R * Cn;
    unsigned short* D = dst + (size_t)g * R * Cn;
    const int c  = threadIdx.x & 31;
    const int r4 = threadIdx.x >> 5;
    for (int i = 0; i < 4; i++) {
        const int r = r4 * 4 + i;
        t[r][c] = f2bf(S[(size_t)(r0 + r) * Cn + c0 + c]);
    }
    __syncthreads();
    for (int i = 0; i < 4; i++) {
        const int r = r4 * 4 + i;
        D[(size_t)(c0 + r) * R + r0 + c] = t[c][r];
    }
}

// ---------------------------------------------------------------------------
// In-place LayerNorm + exact GELU over HID=2048 bf16 rows. One block per row.
// ---------------------------------------------------------------------------
__global__ __launch_bounds__(256) void ln_gelu_kernel(
    unsigned short* __restrict__ hb,
    const float* __restrict__ ln_g, const float* __restrict__ ln_b)
{
    __shared__ float r1[256];
    __shared__ float r2[256];
    const int row = blockIdx.x;          // g*B + b
    const int g   = row >> 12;           // /4096
    const int tid = threadIdx.x;
    unsigned short* rp = hb + (size_t)row * HID_D;

    uint4 v = *(uint4*)(rp + tid * 8);
    float x[8];
    x[0] = bf2f(v.x & 0xffff); x[1] = bf2f(v.x >> 16);
    x[2] = bf2f(v.y & 0xffff); x[3] = bf2f(v.y >> 16);
    x[4] = bf2f(v.z & 0xffff); x[5] = bf2f(v.z >> 16);
    x[6] = bf2f(v.w & 0xffff); x[7] = bf2f(v.w >> 16);

    float s = 0.f, sq = 0.f;
    for (int e = 0; e < 8; e++) { s += x[e]; sq += x[e] * x[e]; }
    r1[tid] = s; r2[tid] = sq;
    __syncthreads();
    for (int off = 128; off; off >>= 1) {
        if (tid < off) { r1[tid] += r1[tid + off]; r2[tid] += r2[tid + off]; }
        __syncthreads();
    }
    const float mu  = r1[0] * (1.0f / HID_D);
    const float var = r2[0] * (1.0f / HID_D) - mu * mu;
    const float inv = rsqrtf(var + 1e-5f);
    const float* lg = ln_g + (size_t)g * HID_D;
    const float* lb = ln_b + (size_t)g * HID_D;

    unsigned int o[4];
    for (int p = 0; p < 4; p++) {
        unsigned short lo = 0, hi = 0;
        for (int q = 0; q < 2; q++) {
            const int e = p * 2 + q;
            const int n = tid * 8 + e;
            float y = (x[e] - mu) * inv * lg[n] + lb[n];
            y = 0.5f * y * (1.0f + erff(y * 0.70710678118654752f));
            if (q == 0) lo = f2bf(y); else hi = f2bf(y);
        }
        o[p] = (unsigned int)lo | ((unsigned int)hi << 16);
    }
    *(uint4*)(rp + tid * 8) = make_uint4(o[0], o[1], o[2], o[3]);
}

// ---------------------------------------------------------------------------
// L2-normalize q rows (PROJ=256) f32 -> bf16. One wave per row, 8 rows/block.
// ---------------------------------------------------------------------------
__global__ __launch_bounds__(256) void qnorm_kernel(
    const float* __restrict__ qraw, unsigned short* __restrict__ qb)
{
    const int row  = blockIdx.x * 8 + (threadIdx.x >> 5);
    const int lane = threadIdx.x & 31;
    const float* rp = qraw + (size_t)row * PROJ_D + lane * 8;
    const float4 a = ((const float4*)rp)[0];
    const float4 c = ((const float4*)rp)[1];
    float s = a.x*a.x + a.y*a.y + a.z*a.z + a.w*a.w
            + c.x*c.x + c.y*c.y + c.z*c.z + c.w*c.w;
    for (int off = 16; off; off >>= 1) s += __shfl_xor(s, off, 32);
    const float invn = 1.0f / fmaxf(sqrtf(s), 1e-12f);
    const unsigned o0 = (unsigned)f2bf(a.x*invn) | ((unsigned)f2bf(a.y*invn) << 16);
    const unsigned o1 = (unsigned)f2bf(a.z*invn) | ((unsigned)f2bf(a.w*invn) << 16);
    const unsigned o2 = (unsigned)f2bf(c.x*invn) | ((unsigned)f2bf(c.y*invn) << 16);
    const unsigned o3 = (unsigned)f2bf(c.z*invn) | ((unsigned)f2bf(c.w*invn) << 16);
    *(uint4*)(qb + (size_t)row * PROJ_D + lane * 8) = make_uint4(o0, o1, o2, o3);
}

// ---------------------------------------------------------------------------
// factor[g,v] = min(exp(logit_scale[g]),100) / max(||Wv[g,v,:]||, 1e-12)
// ---------------------------------------------------------------------------
__global__ __launch_bounds__(256) void wvfactor_kernel(
    const float* __restrict__ Wv, const float* __restrict__ logit_scale,
    float* __restrict__ factor)
{
    const int row  = blockIdx.x * 8 + (threadIdx.x >> 5);   // g*CHUNK + v
    const int lane = threadIdx.x & 31;
    const float* rp = Wv + (size_t)row * PROJ_D + lane * 8;
    const float4 a = ((const float4*)rp)[0];
    const float4 c = ((const float4*)rp)[1];
    float s = a.x*a.x + a.y*a.y + a.z*a.z + a.w*a.w
            + c.x*c.x + c.y*c.y + c.z*c.z + c.w*c.w;
    for (int off = 16; off; off >>= 1) s += __shfl_xor(s, off, 32);
    if (lane == 0) {
        const int g = row >> 12;    // /CHUNK
        const float sc = fminf(expf(logit_scale[g]), 100.0f);
        factor[row] = sc / fmaxf(sqrtf(s), 1e-12f);
    }
}

extern "C" void kernel_launch(void* const* d_in, const int* in_sizes, int n_in,
                              void* d_out, int out_size, void* d_ws, size_t ws_size,
                              hipStream_t stream) {
    const float* z    = (const float*)d_in[0];
    const float* W1   = (const float*)d_in[1];
    const float* b1   = (const float*)d_in[2];
    const float* ln_g = (const float*)d_in[3];
    const float* ln_b = (const float*)d_in[4];
    const float* W2   = (const float*)d_in[5];
    const float* b2   = (const float*)d_in[6];
    const float* Wv   = (const float*)d_in[7];
    const float* bv   = (const float*)d_in[8];
    const float* lsc  = (const float*)d_in[9];
    float* out = (float*)d_out;

    // workspace layout (bytes):
    char* ws = (char*)d_ws;
    unsigned short* hb   = (unsigned short*)ws;                      // 268,435,456
    float*          qraw = (float*)(ws + 268435456ull);              //  67,108,864
    unsigned short* qb   = (unsigned short*)(ws + 335544320ull);     //  33,554,432
    float*          fac  = (float*)(ws + 369098752ull);              //     262,144
    unsigned short* zb   = (unsigned short*)(ws + 369360896ull);     //   8,388,608
    unsigned short* W1t  = (unsigned short*)(ws + 377749504ull);     //  67,108,864
    unsigned short* W2t  = (unsigned short*)(ws + 444858368ull);     //  16,777,216
    unsigned short* Wvb  = (unsigned short*)(ws + 461635584ull);     //  33,554,432

    const dim3 blk(256);

    // --- pre-pass: bf16 conversions (+ weight transposes to [N,K]) ---
    cvt_bf16_kernel<<<dim3((B_DIM * IN_D) / 1024), blk, 0, stream>>>(z, zb);
    transpose_bf16_kernel<<<dim3(IN_D / 32, HID_D / 32, G_DIM), blk, 0, stream>>>(W1, W1t, IN_D, HID_D);
    transpose_bf16_kernel<<<dim3(HID_D / 32, PROJ_D / 32, G_DIM), blk, 0, stream>>>(W2, W2t, HID_D, PROJ_D);
    cvt_bf16_kernel<<<dim3((G_DIM * CHUNK_D * PROJ_D) / 1024), blk, 0, stream>>>(Wv, Wvb);

    // GEMM1: hb[g] = bf16(zb @ W1t[g]^T + b1[g])
    wmma_gemm_kernel<0><<<dim3(HID_D / 128, B_DIM / 128, G_DIM), blk, 0, stream>>>(
        zb, W1t, hb, b1, nullptr,
        IN_D, HID_D, HID_D,
        0L, (long)IN_D * HID_D, (long)B_DIM * HID_D);

    // LayerNorm + GELU, in place on hb
    ln_gelu_kernel<<<dim3(G_DIM * B_DIM), blk, 0, stream>>>(hb, ln_g, ln_b);

    // GEMM2: qraw[g] = hb[g] @ W2t[g]^T + b2[g]
    wmma_gemm_kernel<1><<<dim3(PROJ_D / 128, B_DIM / 128, G_DIM), blk, 0, stream>>>(
        hb, W2t, qraw, b2, nullptr,
        HID_D, PROJ_D, PROJ_D,
        (long)B_DIM * HID_D, (long)HID_D * PROJ_D, (long)B_DIM * PROJ_D);

    // q row L2-normalize -> bf16
    qnorm_kernel<<<dim3(G_DIM * B_DIM / 8), blk, 0, stream>>>(qraw, qb);

    // per-row scale/norm factor for Wv (from f32 Wv)
    wvfactor_kernel<<<dim3(G_DIM * CHUNK_D / 8), blk, 0, stream>>>(Wv, lsc, fac);

    // GEMM3: out[b, g*CHUNK+v] = (qb[g] @ Wvb[g]^T) * fac[g,v] + bv[g,v]
    wmma_gemm_kernel<2><<<dim3(CHUNK_D / 128, B_DIM / 128, G_DIM), blk, 0, stream>>>(
        qb, Wvb, out, fac, bv,
        PROJ_D, G_DIM * CHUNK_D, CHUNK_D,
        (long)B_DIM * PROJ_D, (long)CHUNK_D * PROJ_D, (long)CHUNK_D);
}
